// BiBoAttention_75239237091861
// MI455X (gfx1250) — compile-verified, matrix operand order checked by reading, not run on hardware
//
#include <hip/hip_runtime.h>

// Problem constants (match reference)
constexpr int kB = 2, kS = 2048, kD = 2048, kH = 16, kHD = 128;

typedef __attribute__((ext_vector_type(16))) __bf16 v16bf;
typedef __attribute__((ext_vector_type(8)))  float  v8f;
typedef __attribute__((ext_vector_type(4)))  unsigned uint4v;
typedef __attribute__((ext_vector_type(8)))  int      int8v;
typedef __attribute__((ext_vector_type(4)))  int      int4v;

union Frag { v16bf b; unsigned int u[8]; };

__device__ __forceinline__ unsigned short f2bf(float f) {
  unsigned int u = __builtin_bit_cast(unsigned int, f);
  u += 0x7fffu + ((u >> 16) & 1u);          // round-to-nearest-even
  return (unsigned short)(u >> 16);
}
__device__ __forceinline__ float bf2f(unsigned short h) {
  unsigned int u = ((unsigned int)h) << 16;
  return __builtin_bit_cast(float, u);
}

// K-pair base index for 16-bit WMMA A/B fragments (ISA 7.12.2):
// VGPR 0..3 hold K = {0..7 | 8..15}, VGPR 4..7 hold K = {16..23 | 24..31},
// low/high half selected by lane[4].
__device__ __forceinline__ int kpat(int v, int lane) {
  int base = (v < 4) ? ((lane & 16) ? 8 : 0) : ((lane & 16) ? 24 : 16);
  return base + 2 * (v & 3);
}

__device__ __forceinline__ v8f wmma_bf16(const Frag& a, const Frag& bb, v8f c) {
  return __builtin_amdgcn_wmma_f32_16x16x32_bf16(
      false, a.b, false, bb.b, (short)0, c, false, false);
}

// ---------------------------------------------------------------------------
// TDM: 2D tile DMA global -> LDS (CDNA5 ISA).
// Elements are 2 bytes (bf16). Groups 2/3 zeroed (2D tensor, no iterate).
// pad_interval code: interval = 2<<code dwords; pad_amount code: (code+1) dwords.
// ---------------------------------------------------------------------------
__device__ __forceinline__ void tdm_load_2d(
    unsigned lds_off, const void* gaddr,
    unsigned tensor_d0, unsigned tensor_d1, unsigned d0_stride,
    unsigned tile_d0, unsigned tile_d1,
    unsigned pad_interval_code, unsigned pad_amount_code)
{
  const unsigned long long ga = (unsigned long long)gaddr;
  uint4v g0;
  g0[0] = 1u;                                         // count=1, user D#
  g0[1] = lds_off;                                    // lds_addr
  g0[2] = (unsigned)ga;                               // global_addr[31:0]
  g0[3] = (unsigned)((ga >> 32) & 0x01FFFFFFull)      // global_addr[56:32]
        | 0x80000000u;                                // type=2 ("image")
  int8v g1;
  g1[0] = (int)((1u << 16)                            // data_size = 2 bytes
        | (1u << 20)                                  // pad_enable
        | (pad_interval_code << 22)
        | (pad_amount_code << 25));
  g1[1] = (int)((tensor_d0 & 0xFFFFu) << 16);         // dim0[15:0]
  g1[2] = (int)(((tensor_d0 >> 16) & 0xFFFFu) | ((tensor_d1 & 0xFFFFu) << 16));
  g1[3] = (int)(((tensor_d1 >> 16) & 0xFFFFu) | ((tile_d0 & 0xFFFFu) << 16));
  g1[4] = (int)(tile_d1 & 0xFFFFu);                   // tile_dim1; tile_dim2=0
  g1[5] = (int)d0_stride;                             // dim0_stride[31:0]
  g1[6] = 0;                                          // dim0_stride[47:32]
  g1[7] = 0;
  const int4v gz = {0, 0, 0, 0};
#if __clang_major__ >= 23
  const int8v gz8 = {0, 0, 0, 0, 0, 0, 0, 0};
  __builtin_amdgcn_tensor_load_to_lds(g0, g1, gz, gz, gz8, 0);
#else
  __builtin_amdgcn_tensor_load_to_lds(g0, g1, gz, gz, 0);
#endif
}

// ---------------------------------------------------------------------------
// GEMM: C[M,N] = A[M,K] (f32 rm) x W[K,N] (f32 rm). Block tile 128x128x32,
// 8 waves, each wave owns a 32x64 sub-tile -> 8 WMMA per k-step.
// mode 0: Cf = f32 row-major [M,N]
// mode 1: Cb = bf16 scatter to [B,H,S,HD]
// mode 2: Cb = bf16 scatter to [B,H,HD,S]
// ---------------------------------------------------------------------------
#define GBM 128
#define GBN 128
#define GBK 32

__global__ __launch_bounds__(256) void gemm_kernel(
    const float* __restrict__ A, const float* __restrict__ W,
    float* __restrict__ Cf, unsigned short* __restrict__ Cb,
    int M, int N, int K, int mode)
{
  __shared__ unsigned short As[GBM][GBK + 2];   // 17-dword row stride
  __shared__ unsigned short BsT[GBN][GBK + 2];  // W tile stored N-major

  const int tid  = threadIdx.x;
  const int lane = tid & 31;
  const int wave = tid >> 5;
  const int wr = wave >> 1;                     // 0..3: 32-row group
  const int wc = wave & 1;                      // 0..1: 64-col group
  const int bm = blockIdx.y * GBM;
  const int bn = blockIdx.x * GBN;

  v8f acc[2][4] = {};

  for (int k0 = 0; k0 < K; k0 += GBK) {
    // stage A tile (128x32), f32 -> bf16
#pragma unroll
    for (int i = 0; i < (GBM * GBK) / 256; ++i) {
      int e = tid + i * 256;
      int r = e >> 5, c = e & 31;
      As[r][c] = f2bf(A[(size_t)(bm + r) * K + (k0 + c)]);
    }
    // stage W tile (32x128) transposed, f32 -> bf16
#pragma unroll
    for (int i = 0; i < (GBK * GBN) / 256; ++i) {
      int e = tid + i * 256;
      int r = e >> 7, c = e & 127;              // r = k, c = n
      BsT[c][r] = f2bf(W[(size_t)(k0 + r) * N + (bn + c)]);
    }
    if (k0 + GBK < K) {                         // prefetch next k-tile
      __builtin_prefetch(&A[(size_t)(bm + (tid >> 5)) * K + (k0 + GBK) + (tid & 31)], 0, 0);
      __builtin_prefetch(&W[(size_t)(k0 + GBK + (tid >> 7)) * N + bn + (tid & 127)], 0, 0);
    }
    __syncthreads();

    Frag af[2];
#pragma unroll
    for (int mi = 0; mi < 2; ++mi) {
      const int ar = wr * 32 + mi * 16 + (lane & 15);
#pragma unroll
      for (int v = 0; v < 8; ++v)
        af[mi].u[v] = *(const unsigned int*)&As[ar][kpat(v, lane)];
    }
#pragma unroll
    for (int nt = 0; nt < 4; ++nt) {
      Frag bfr;
      const int bc = wc * 64 + nt * 16 + (lane & 15);
#pragma unroll
      for (int v = 0; v < 8; ++v)
        bfr.u[v] = *(const unsigned int*)&BsT[bc][kpat(v, lane)];
#pragma unroll
      for (int mi = 0; mi < 2; ++mi)
        acc[mi][nt] = wmma_bf16(af[mi], bfr, acc[mi][nt]);
    }
    __syncthreads();
  }

  // epilogue: C fragment row = r + 8*(lane>=16), col = lane&15 per 16-tile
#pragma unroll
  for (int mi = 0; mi < 2; ++mi) {
#pragma unroll
    for (int nt = 0; nt < 4; ++nt) {
#pragma unroll
      for (int r = 0; r < 8; ++r) {
        const int row = bm + wr * 32 + mi * 16 + r + ((lane >> 4) << 3);
        const int col = bn + wc * 64 + nt * 16 + (lane & 15);
        const float val = acc[mi][nt][r];
        if (mode == 0) {
          Cf[(size_t)row * N + col] = val;
        } else {
          const int b = row / kS, s = row % kS;
          const int h = col / kHD, hd = col % kHD;
          if (mode == 1)
            Cb[((size_t)(b * kH + h) * kS + s) * kHD + hd] = f2bf(val);
          else
            Cb[((size_t)(b * kH + h) * kHD + hd) * kS + s] = f2bf(val);
        }
      }
    }
  }
}

// ---------------------------------------------------------------------------
// RoPE in place on bf16 Q and K laid out [B,H,S,HD]; pairs (i, i+64).
// ---------------------------------------------------------------------------
__global__ __launch_bounds__(256) void rope_kernel(
    unsigned short* __restrict__ Qb, unsigned short* __restrict__ Kb)
{
  const int idx = blockIdx.x * 256 + threadIdx.x;   // B*H*S*64 threads
  const int i  = idx & 63;
  const int s  = (idx >> 6) & (kS - 1);
  const int bh = idx >> 17;                         // /(64*2048)
  const size_t base = ((size_t)bh * kS + s) * kHD;

  const float inv_freq = __expf(-(float)i * (9.2103403719761840f / 64.0f));
  const float fr = (float)s * inv_freq;
  const float c = cosf(fr), sn = sinf(fr);

  {
    float x1 = bf2f(Qb[base + i]);
    float x2 = bf2f(Qb[base + i + 64]);
    Qb[base + i]      = f2bf(x1 * c - x2 * sn);
    Qb[base + i + 64] = f2bf(x2 * c + x1 * sn);
  }
  {
    float x1 = bf2f(Kb[base + i]);
    float x2 = bf2f(Kb[base + i + 64]);
    Kb[base + i]      = f2bf(x1 * c - x2 * sn);
    Kb[base + i + 64] = f2bf(x2 * c + x1 * sn);
  }
}

// ---------------------------------------------------------------------------
// Flash attention, block-cooperative: one block = one (b,h) x 128 queries
// (8 waves x 16-query tiles). K/V tiles DMA'd to LDS by the Tensor Data
// Mover (wave 0 issues, waits TENSORcnt, workgroup barrier publishes).
// TDM pad fields reproduce the bank-conflict-avoiding row padding.
// Q,K bf16 [B,H,S,HD]; V bf16 transposed [B,H,HD,S]; Out f32 [B,S,D].
// ---------------------------------------------------------------------------
#define KSP 136   // Ks row stride: 128 data + 8 pad (64-dword rows + 4-dword pad)
#define VSP 40    // VTs row stride: 32 data + 8 pad (16-dword rows + 4-dword pad)
#define PSP 34    // P row stride (32 + 2 pad)

__global__ __launch_bounds__(256) void attn_kernel(
    const unsigned short* __restrict__ Qb,
    const unsigned short* __restrict__ Kb,
    const unsigned short* __restrict__ VT,
    float* __restrict__ Out)
{
  __shared__ unsigned short Ks[32][KSP];        // keys x hd      (8.5 KB)
  __shared__ unsigned short VTs[128][VSP];      // hd x keys      (10 KB)
  __shared__ unsigned short plds[8][16 * PSP];  // per-wave P     (8.5 KB)

  const int tid  = threadIdx.x;
  const int lane = tid & 31;
  const int wave = tid >> 5;
  const int qBlocks = kS / 128;                 // 16
  const int bh  = blockIdx.x / qBlocks;
  const int qb0 = (blockIdx.x % qBlocks) * 128;
  const int qbase = qb0 + wave * 16;
  const int b = bh / kH, h = bh % kH;

  const unsigned short* Qh = Qb + (size_t)bh * kS * kHD;
  const unsigned short* Kh = Kb + (size_t)bh * kS * kHD;
  const unsigned short* Vh = VT + (size_t)bh * kHD * kS;

  // LDS byte offsets (generic-pointer low 32 bits == LDS offset, ISA 10.2)
  const unsigned ksOff = (unsigned)(unsigned long long)(const void*)&Ks[0][0];
  const unsigned vtOff = (unsigned)(unsigned long long)(const void*)&VTs[0][0];

  // Q fragments: 16x128 split into 4 (16x32) A-operands
  Frag qf[4];
  const int qr = qbase + (lane & 15);
#pragma unroll
  for (int kf = 0; kf < 4; ++kf)
#pragma unroll
    for (int v = 0; v < 8; ++v)
      qf[kf].u[v] =
          *(const unsigned int*)&Qh[(size_t)qr * kHD + kf * 32 + kpat(v, lane)];

  v8f o[8] = {};
  float mrun[8], lrun[8];
#pragma unroll
  for (int r = 0; r < 8; ++r) { mrun[r] = -1e30f; lrun[r] = 0.0f; }

  const float scale = 0.08838834764831845f;     // 1/sqrt(128)
  unsigned short* myp = &plds[wave][0];
  const int jend = qb0 + 128;

  for (int j = 0; j < jend; j += 32) {
    // ---- TDM staging: K tile 32x128 rows, V^T tile 128x32 rows ----
    if (wave == 0) {
      // K: tensor [S,HD] (dim0=128, stride=128), tile 128x32 rows,
      //    rows are 64 dwords -> pad_interval code 5, pad 4 dwords -> code 3
      tdm_load_2d(ksOff, Kh + (size_t)j * kHD,
                  kHD, kS, kHD, /*tile_d0=*/kHD, /*tile_d1=*/32, 5, 3);
      // V^T: tensor [HD,S] (dim0=2048, stride=2048), tile 32x128 rows,
      //    rows are 16 dwords -> pad_interval code 3, pad 4 dwords -> code 3
      tdm_load_2d(vtOff, Vh + j,
                  kS, kHD, kS, /*tile_d0=*/32, /*tile_d1=*/kHD, 3, 3);
      __builtin_amdgcn_s_wait_tensorcnt(0);
    }
    __syncthreads();                            // publish tiles to all waves

    // ---- S = Q K^T for 32 keys (two 16-key N tiles), K-dim = 128 ----
    v8f s0 = {}, s1 = {};
#pragma unroll
    for (int kf = 0; kf < 4; ++kf) {
      Frag k0f, k1f;
      const int kk0 = lane & 15;
      const int kk1 = 16 + (lane & 15);
#pragma unroll
      for (int v = 0; v < 8; ++v) {
        const int hh = kf * 32 + kpat(v, lane);
        k0f.u[v] = *(const unsigned int*)&Ks[kk0][hh];
        k1f.u[v] = *(const unsigned int*)&Ks[kk1][hh];
      }
      s0 = wmma_bf16(qf[kf], k0f, s0);
      s1 = wmma_bf16(qf[kf], k1f, s1);
    }

    // ---- online softmax; rows in VGPR index, cols across 16-lane groups ----
#pragma unroll
    for (int r = 0; r < 8; ++r) {
      const int row = qbase + r + ((lane >> 4) << 3);
      const int c   = lane & 15;
      float v0 = s0[r] * scale;
      float v1 = s1[r] * scale;
      if (j + c > row)      v0 = -1e30f;        // causal mask
      if (j + 16 + c > row) v1 = -1e30f;
      float mx = fmaxf(v0, v1);
      mx = fmaxf(mx, __shfl_xor(mx, 1, 32));
      mx = fmaxf(mx, __shfl_xor(mx, 2, 32));
      mx = fmaxf(mx, __shfl_xor(mx, 4, 32));
      mx = fmaxf(mx, __shfl_xor(mx, 8, 32));
      const float mnew = fmaxf(mrun[r], mx);
      const float corr = __expf(mrun[r] - mnew);
      mrun[r] = mnew;
      const float p0 = __expf(v0 - mnew);
      const float p1 = __expf(v1 - mnew);
      float rs = p0 + p1;
      rs += __shfl_xor(rs, 1, 32);
      rs += __shfl_xor(rs, 2, 32);
      rs += __shfl_xor(rs, 4, 32);
      rs += __shfl_xor(rs, 8, 32);
      lrun[r] = lrun[r] * corr + rs;
#pragma unroll
      for (int nt = 0; nt < 8; ++nt) o[nt][r] *= corr;
      const int rl = r + ((lane >> 4) << 3);    // C-layout -> LDS (16x32)
      myp[rl * PSP + c]      = f2bf(p0);
      myp[rl * PSP + 16 + c] = f2bf(p1);
    }
    asm volatile("s_wait_dscnt 0" ::: "memory"); // wave-local LDS fence

    // ---- reload P as a 16x32 A-operand ----
    Frag pa;
#pragma unroll
    for (int v = 0; v < 8; ++v)
      pa.u[v] = *(const unsigned int*)&myp[(lane & 15) * PSP + kpat(v, lane)];

    // ---- O += P V ----
#pragma unroll
    for (int nt = 0; nt < 8; ++nt) {
      Frag vf;
      const int hd = nt * 16 + (lane & 15);
#pragma unroll
      for (int v = 0; v < 8; ++v)
        vf.u[v] = *(const unsigned int*)&VTs[hd][kpat(v, lane)];
      o[nt] = wmma_bf16(pa, vf, o[nt]);
    }
    __syncthreads();                            // protect Ks/VTs for next j
  }

  // normalize and write [B,S,D] f32
  float inv[8];
#pragma unroll
  for (int r = 0; r < 8; ++r) inv[r] = 1.0f / lrun[r];
#pragma unroll
  for (int nt = 0; nt < 8; ++nt) {
#pragma unroll
    for (int r = 0; r < 8; ++r) {
      const int q  = qbase + r + ((lane >> 4) << 3);
      const int hd = nt * 16 + (lane & 15);
      Out[(size_t)(b * kS + q) * kD + h * kHD + hd] = o[nt][r] * inv[r];
    }
  }
}

// ---------------------------------------------------------------------------
extern "C" void kernel_launch(void* const* d_in, const int* in_sizes, int n_in,
                              void* d_out, int out_size, void* d_ws, size_t ws_size,
                              hipStream_t stream)
{
  (void)in_sizes; (void)n_in; (void)out_size; (void)ws_size;
  const float* X  = (const float*)d_in[0];
  // d_in[1] = additive causal mask (regenerated analytically in-kernel)
  // d_in[2] = position_ids (== arange(S), used analytically in rope_kernel)
  const float* Wq = (const float*)d_in[3];
  const float* Wk = (const float*)d_in[4];
  const float* Wv = (const float*)d_in[5];
  const float* Wo = (const float*)d_in[6];
  float* out = (float*)d_out;

  const int M = kB * kS;                                  // 4096
  const size_t qkvElems = (size_t)kB * kH * kS * kHD;     // 8,388,608

  unsigned short* Qb = (unsigned short*)d_ws;
  unsigned short* Kb = Qb + qkvElems;
  unsigned short* VT = Kb + qkvElems;
  float* attn = (float*)(VT + qkvElems);                  // f32 [B*S, D]

  dim3 blk(256);
  dim3 ggrid(kD / GBN, M / GBM);                          // 16 x 32

  gemm_kernel<<<ggrid, blk, 0, stream>>>(X, Wq, nullptr, Qb, M, kD, kD, 1);
  gemm_kernel<<<ggrid, blk, 0, stream>>>(X, Wk, nullptr, Kb, M, kD, kD, 1);
  gemm_kernel<<<ggrid, blk, 0, stream>>>(X, Wv, nullptr, VT, M, kD, kD, 2);

  rope_kernel<<<(kB * kH * kS * 64) / 256, blk, 0, stream>>>(Qb, Kb);

  attn_kernel<<<kB * kH * (kS / 128), blk, 0, stream>>>(Qb, Kb, VT, attn);

  gemm_kernel<<<ggrid, blk, 0, stream>>>(attn, Wo, out, nullptr, M, kD, kD, 0);
}
